// GameOutcomePredictor_3324304687518
// MI455X (gfx1250) — matrix operand order repaired and checked by reading, not compiled, not run on hardware
//
#include <hip/hip_runtime.h>
#include <hip/hip_bf16.h>

typedef __attribute__((ext_vector_type(16))) _Float16 v16h;
typedef __attribute__((ext_vector_type(8)))  _Float16 v8h;
typedef __attribute__((ext_vector_type(8)))  float    v8f;

#define N_NODES 100000   // divisible by 16 -> all row tiles full
#define N_EDGES 1200000
#define IN_CH   22
#define KPAD    32
#define HID     64
#define NCLS    3
#define NGRAPH  256

// ---- K-index mapping for 16-bit A/B 16x32 fragments (CDNA5 ISA 7.12.2) ----
// lane L: M(or N) = L&15, half = L>>4; fragment element i (VGPR v = i>>1, pos = i&1):
//   K = (v<4 ? 0 : 16) + 8*half + (v&3)*2 + pos
// => elements 0..7  hold contiguous K in [ 8*half,  8*half+8)
//    elements 8..15 hold contiguous K in [16+8*half, 16+8*half+8)
__device__ __forceinline__ int kmap(int i, int half) {
    int v = i >> 1, pos = i & 1;
    return ((v < 4) ? 0 : 16) + half * 8 + (v & 3) * 2 + pos;
}

__device__ __forceinline__ v16h load_frag32(const _Float16* __restrict__ base, int half) {
    v8h lo = *(const v8h*)(base + 8 * half);        // K [8h, 8h+8)
    v8h hi = *(const v8h*)(base + 16 + 8 * half);   // K [16+8h, 16+8h+8)
    return __builtin_shufflevector(lo, hi, 0, 1, 2, 3, 4, 5, 6, 7,
                                   8, 9, 10, 11, 12, 13, 14, 15);
}

// -------------------- utility --------------------
__global__ void fill_zero_kernel(float* __restrict__ p, int n) {
    int t = blockIdx.x * blockDim.x + threadIdx.x;
    if (t < n) p[t] = 0.0f;
}

// -------------------- degree / normalization --------------------
__global__ void deg_accum_kernel(const int* __restrict__ ei,
                                 const float* __restrict__ w,
                                 float* __restrict__ deg, int E) {
    int e = blockIdx.x * blockDim.x + threadIdx.x;
    if (e >= E) return;
    atomicAdd(&deg[ei[E + e]], w[e]);   // col = edge_index[1][e]
}

__global__ void finalize_dinv_kernel(float* __restrict__ dinv, int n) {
    int i = blockIdx.x * blockDim.x + threadIdx.x;
    if (i >= n) return;
    dinv[i] = __frsqrt_rn(dinv[i] + 1.0f);  // + self-loop weight 1 => deg >= 1
}

__global__ void compute_norm_kernel(const int* __restrict__ ei,
                                    const float* __restrict__ w,
                                    const float* __restrict__ dinv,
                                    float* __restrict__ norm, int E) {
    int e = blockIdx.x * blockDim.x + threadIdx.x;
    if (e >= E) return;
    norm[e] = dinv[ei[e]] * w[e] * dinv[ei[E + e]];
}

// -------------------- fragment prep --------------------
// x [N,22] f32 -> xh [N,32] f16, zero padded
__global__ void cvt_x_kernel(const float* __restrict__ x,
                             _Float16* __restrict__ xh, int n) {
    int t = blockIdx.x * blockDim.x + threadIdx.x;
    if (t >= n * KPAD) return;
    int i = t >> 5, k = t & 31;
    xh[t] = (k < IN_CH) ? (_Float16)x[i * IN_CH + k] : (_Float16)0.0f;
}

// W1 [22,64] -> fragment order: wf[(c*32+lane)*16 + i], zero padded K 22..31
__global__ void prep_w1_kernel(const float* __restrict__ W,
                               _Float16* __restrict__ wf) {
    int t = blockIdx.x * blockDim.x + threadIdx.x;
    if (t >= 4 * 32 * 16) return;
    int i = t & 15, lane = (t >> 4) & 31, c = t >> 9;
    int k = kmap(i, lane >> 4);
    int col = c * 16 + (lane & 15);
    wf[t] = (k < IN_CH) ? (_Float16)W[k * HID + col] : (_Float16)0.0f;
}

// W2 [64,64] -> fragment order: wf[((c*2+ks)*32+lane)*16 + i]
__global__ void prep_w2_kernel(const float* __restrict__ W,
                               _Float16* __restrict__ wf) {
    int t = blockIdx.x * blockDim.x + threadIdx.x;
    if (t >= 4 * 2 * 32 * 16) return;
    int i = t & 15, lane = (t >> 4) & 31, ks = (t >> 9) & 1, c = t >> 10;
    int k = 32 * ks + kmap(i, lane >> 4);
    int col = c * 16 + (lane & 15);
    wf[t] = (_Float16)W[k * HID + col];
}

// -------------------- WMMA GEMM1: xh[N,32] @ W1 -> out[N,64] --------------------
__global__ void gemm1_wmma_kernel(const _Float16* __restrict__ xh,
                                  const _Float16* __restrict__ wf,
                                  float* __restrict__ out, int ntiles) {
    int wave = (blockIdx.x * blockDim.x + threadIdx.x) >> 5;
    int lane = threadIdx.x & 31;
    if (wave >= ntiles) return;            // wave-uniform: EXEC stays all-1s
    int half = lane >> 4;
    int row  = wave * 16 + (lane & 15);
    v16h a = load_frag32(xh + (size_t)row * KPAD, half);
#pragma unroll
    for (int c = 0; c < 4; ++c) {
        v16h b = *(const v16h*)(wf + ((size_t)(c * 32 + lane)) * 16);
        v8f acc = {};
        acc = __builtin_amdgcn_wmma_f32_16x16x32_f16(false, a, false, b,
                                                     (short)0, acc, false, false);
        int col = c * 16 + (lane & 15);
        float* o = out + (size_t)(wave * 16 + half * 8) * HID + col;
#pragma unroll
        for (int r = 0; r < 8; ++r) o[r * HID] = acc[r];
    }
}

// -------------------- WMMA GEMM2: h16[N,64] @ W2 -> out[N,64] --------------------
__global__ void gemm2_wmma_kernel(const _Float16* __restrict__ h16,
                                  const _Float16* __restrict__ wf,
                                  float* __restrict__ out, int ntiles) {
    int wave = (blockIdx.x * blockDim.x + threadIdx.x) >> 5;
    int lane = threadIdx.x & 31;
    if (wave >= ntiles) return;
    int half = lane >> 4;
    int row  = wave * 16 + (lane & 15);
    const _Float16* hr = h16 + (size_t)row * HID;
    v16h a0 = load_frag32(hr, half);        // K 0..31
    v16h a1 = load_frag32(hr + 32, half);   // K 32..63
#pragma unroll
    for (int c = 0; c < 4; ++c) {
        v16h b0 = *(const v16h*)(wf + ((size_t)((c * 2 + 0) * 32 + lane)) * 16);
        v16h b1 = *(const v16h*)(wf + ((size_t)((c * 2 + 1) * 32 + lane)) * 16);
        v8f acc = {};
        acc = __builtin_amdgcn_wmma_f32_16x16x32_f16(false, a0, false, b0,
                                                     (short)0, acc, false, false);
        acc = __builtin_amdgcn_wmma_f32_16x16x32_f16(false, a1, false, b1,
                                                     (short)0, acc, false, false);
        int col = c * 16 + (lane & 15);
        float* o = out + (size_t)(wave * 16 + half * 8) * HID + col;
#pragma unroll
        for (int r = 0; r < 8; ++r) o[r * HID] = acc[r];
    }
}

// -------------------- edge message scatter (fp32 atomics, L2-resident) --------
__global__ void edge_scatter_kernel(const int* __restrict__ ei,
                                    const float* __restrict__ norm,
                                    const float* __restrict__ h,
                                    float* __restrict__ agg, int E) {
    int t = blockIdx.x * blockDim.x + threadIdx.x;
    int e = t >> 4;
    if (e >= E) return;
    int c = (t & 15) << 2;
    float w = norm[e];
    const float4 hv = *(const float4*)(h + (size_t)ei[e] * HID + c);
    float* dst = agg + (size_t)ei[E + e] * HID + c;
    atomicAdd(dst + 0, w * hv.x);
    atomicAdd(dst + 1, w * hv.y);
    atomicAdd(dst + 2, w * hv.z);
    atomicAdd(dst + 3, w * hv.w);
}

// -------------------- finish: self-loop + bias + relu (+ optional f16 copy) ----
__global__ void finish_conv_kernel(const float* __restrict__ agg,
                                   const float* __restrict__ h,
                                   const float* __restrict__ dinv,
                                   const float* __restrict__ bias,
                                   float* __restrict__ out,
                                   _Float16* __restrict__ out16, int n) {
    int t = blockIdx.x * blockDim.x + threadIdx.x;
    int i = t >> 4;
    if (i >= n) return;
    int c = (t & 15) << 2;
    float s = dinv[i] * dinv[i];           // self-loop norm: dinv * 1 * dinv
    float4 a  = *(const float4*)(agg + (size_t)i * HID + c);
    float4 hv = *(const float4*)(h + (size_t)i * HID + c);
    float4 o;
    o.x = fmaxf(a.x + s * hv.x + bias[c + 0], 0.0f);
    o.y = fmaxf(a.y + s * hv.y + bias[c + 1], 0.0f);
    o.z = fmaxf(a.z + s * hv.z + bias[c + 2], 0.0f);
    o.w = fmaxf(a.w + s * hv.w + bias[c + 3], 0.0f);
    *(float4*)(out + (size_t)i * HID + c) = o;
    if (out16) {                            // uniform branch
        union { _Float16 f[4]; uint2 u; } pk;
        pk.f[0] = (_Float16)o.x; pk.f[1] = (_Float16)o.y;
        pk.f[2] = (_Float16)o.z; pk.f[3] = (_Float16)o.w;
        *(uint2*)(out16 + (size_t)i * HID + c) = pk.u;
    }
}

// -------------------- global mean pool (atomic accumulate) --------------------
__global__ void pool_accum_kernel(const float* __restrict__ h,
                                  const int* __restrict__ batch,
                                  float* __restrict__ sums,
                                  float* __restrict__ cnt, int n) {
    int t = blockIdx.x * blockDim.x + threadIdx.x;
    int i = t >> 4;
    if (i >= n) return;
    int c = (t & 15) << 2;
    int g = batch[i];
    float4 hv = *(const float4*)(h + (size_t)i * HID + c);
    float* dst = sums + (size_t)g * HID + c;
    atomicAdd(dst + 0, hv.x);
    atomicAdd(dst + 1, hv.y);
    atomicAdd(dst + 2, hv.z);
    atomicAdd(dst + 3, hv.w);
    if ((t & 15) == 0) atomicAdd(&cnt[g], 1.0f);
}

// -------------------- tiny MLP head --------------------
__global__ void head1_kernel(const float* __restrict__ sums,
                             const float* __restrict__ cnt,
                             const float* __restrict__ fw1,
                             const float* __restrict__ fb1,
                             float* __restrict__ fc1) {
    int t = blockIdx.x * blockDim.x + threadIdx.x;
    if (t >= NGRAPH * HID) return;
    int g = t >> 6, j = t & 63;
    float inv = 1.0f / fmaxf(cnt[g], 1.0f);
    float acc = fb1[j];
#pragma unroll 8
    for (int k = 0; k < HID; ++k)
        acc += sums[g * HID + k] * inv * fw1[k * HID + j];
    fc1[g * HID + j] = fmaxf(acc, 0.0f);
}

__global__ void head2_kernel(const float* __restrict__ fc1,
                             const float* __restrict__ fw2,
                             const float* __restrict__ fb2,
                             float* __restrict__ out) {
    int t = blockIdx.x * blockDim.x + threadIdx.x;
    if (t >= NGRAPH * NCLS) return;
    int g = t / NCLS, k = t % NCLS;
    float acc = fb2[k];
#pragma unroll 8
    for (int j = 0; j < HID; ++j)
        acc += fc1[g * HID + j] * fw2[j * NCLS + k];
    out[g * NCLS + k] = acc;
}

extern "C" void kernel_launch(void* const* d_in, const int* in_sizes, int n_in,
                              void* d_out, int out_size, void* d_ws, size_t ws_size,
                              hipStream_t stream) {
    const float* x     = (const float*)d_in[0];
    const int*   ei    = (const int*)d_in[1];
    const float* ew    = (const float*)d_in[2];
    const int*   batch = (const int*)d_in[3];
    const float* W1    = (const float*)d_in[4];
    const float* b1    = (const float*)d_in[5];
    const float* W2    = (const float*)d_in[6];
    const float* b2    = (const float*)d_in[7];
    const float* fw1   = (const float*)d_in[8];
    const float* fb1   = (const float*)d_in[9];
    const float* fw2   = (const float*)d_in[10];
    const float* fb2   = (const float*)d_in[11];
    float* out = (float*)d_out;

    const int N = N_NODES, E = N_EDGES;

    // workspace layout (all section offsets multiples of 32 bytes)
    float* ws    = (float*)d_ws;
    float* dinv  = ws;                         // N floats (also deg accumulator)
    float* norm  = dinv + N;                   // E floats
    float* hA    = norm + E;                   // N*64 floats
    float* hB    = hA + (size_t)N * HID;       // N*64 floats
    _Float16* xh  = (_Float16*)(hB + (size_t)N * HID);  // N*32 halfs
    _Float16* h16 = xh + (size_t)N * KPAD;              // N*64 halfs
    _Float16* w1f = h16 + (size_t)N * HID;              // 2048 halfs
    _Float16* w2f = w1f + 4 * 32 * 16;                  // 4096 halfs
    float* sums  = (float*)(w2f + 8 * 32 * 16);         // G*64 floats
    float* cnt   = sums + NGRAPH * HID;                 // G floats
    float* fc1   = cnt + NGRAPH;                        // G*64 floats

    const int B = 256;
    auto blocks = [](long long work, int b) { return (int)((work + b - 1) / b); };

    // --- normalization (shared by both conv layers) ---
    fill_zero_kernel<<<blocks(N, B), B, 0, stream>>>(dinv, N);
    deg_accum_kernel<<<blocks(E, B), B, 0, stream>>>(ei, ew, dinv, E);
    finalize_dinv_kernel<<<blocks(N, B), B, 0, stream>>>(dinv, N);
    compute_norm_kernel<<<blocks(E, B), B, 0, stream>>>(ei, ew, dinv, norm, E);

    // --- fragment prep ---
    cvt_x_kernel<<<blocks((long long)N * KPAD, B), B, 0, stream>>>(x, xh, N);
    prep_w1_kernel<<<blocks(4 * 32 * 16, B), B, 0, stream>>>(W1, w1f);
    prep_w2_kernel<<<blocks(8 * 32 * 16, B), B, 0, stream>>>(W2, w2f);

    int ntiles     = N >> 4;                  // N % 16 == 0
    int gemmBlocks = blocks((long long)ntiles * 32, B);

    // --- conv1: hA = x@W1 ; hB = edge agg ; H1 -> hA (f32) + h16 (f16) ---
    gemm1_wmma_kernel<<<gemmBlocks, B, 0, stream>>>(xh, w1f, hA, ntiles);
    fill_zero_kernel<<<blocks((long long)N * HID, B), B, 0, stream>>>(hB, N * HID);
    edge_scatter_kernel<<<blocks((long long)E * 16, B), B, 0, stream>>>(ei, norm, hA, hB, E);
    finish_conv_kernel<<<blocks((long long)N * 16, B), B, 0, stream>>>(hB, hA, dinv, b1, hA, h16, N);

    // --- conv2: hB = H1@W2 ; hA = edge agg ; H2 -> hB ---
    gemm2_wmma_kernel<<<gemmBlocks, B, 0, stream>>>(h16, w2f, hB, ntiles);
    fill_zero_kernel<<<blocks((long long)N * HID, B), B, 0, stream>>>(hA, N * HID);
    edge_scatter_kernel<<<blocks((long long)E * 16, B), B, 0, stream>>>(ei, norm, hB, hA, E);
    finish_conv_kernel<<<blocks((long long)N * 16, B), B, 0, stream>>>(hA, hB, dinv, b2, hB, (_Float16*)nullptr, N);

    // --- mean pool + MLP head ---
    fill_zero_kernel<<<blocks(NGRAPH * (HID + 1), B), B, 0, stream>>>(sums, NGRAPH * (HID + 1));
    pool_accum_kernel<<<blocks((long long)N * 16, B), B, 0, stream>>>(hB, batch, sums, cnt, N);
    head1_kernel<<<blocks(NGRAPH * HID, B), B, 0, stream>>>(sums, cnt, fw1, fb1, fc1);
    head2_kernel<<<blocks(NGRAPH * NCLS, B), B, 0, stream>>>(fc1, fw2, fb2, out);
}
